// IDS4TokenClassifier_55516747268649
// MI455X (gfx1250) — matrix operand order repaired and checked by reading, not compiled, not run on hardware
//
#include <hip/hip_runtime.h>
#include <hip/hip_bf16.h>
#include <math.h>

// ---------------------------------------------------------------------------
// IDS4 token classifier for MI455X (gfx1250, wave32, WMMA).
// B=16 L=512 DM=512 DS=64 NL=3 V=1024.
// All dense math via v_wmma_f32_16x16x32_bf16 (bf16 in, f32 accumulate);
// async global->LDS staging (ASYNCcnt) in the scan kernels.
// ---------------------------------------------------------------------------

typedef __bf16 bf16_t;
typedef __attribute__((ext_vector_type(2)))  __bf16 bf16x2;
typedef __attribute__((ext_vector_type(16))) __bf16 v16bf;
typedef __attribute__((ext_vector_type(8)))  float  v8f;

#define CB 16
#define CL 512
#define CDM 512
#define CDS 64
#define CNL 3
#define CV 1024
#define NCHUNK 16
#define TCHUNK 32
#define ROWS (CB*CL)     // 8192
#define LNEPS 1e-5f
#define MB (1024ull*1024ull)

__device__ __forceinline__ float gelu_f(float x) {
    return 0.5f * x * (1.0f + erff(x * 0.70710678118654752440f));
}

// --- WMMA fragment loaders (contiguous k = 16*(lane/16) + s, wave32) --------------
__device__ __forceinline__ v16bf lds_load_afrag(const bf16_t* Ls, int ldk,
                                                int m0, int k0, int lane) {
    int g = (lane >> 4) & 1, m = lane & 15;
    const bf16_t* row = Ls + (size_t)(m0 + m) * ldk + k0;
    v16bf a;
#pragma unroll
    for (int r = 0; r < 8; ++r) {
        int kk = (g << 4) + (r << 1);
        bf16x2 p = *(const bf16x2*)(row + kk);
        a[2 * r]     = p.x;
        a[2 * r + 1] = p.y;
    }
    return a;
}
// B-frag from n-major LDS tile Bs[n][k]: 16 consecutive k per lane.
__device__ __forceinline__ v16bf lds_load_bfrag(const bf16_t* Bs, int ldk,
                                                int n0, int k0, int lane) {
    int g = (lane >> 4) & 1, n = lane & 15;
    return *(const v16bf*)(Bs + (size_t)(n0 + n) * ldk + k0 + g * 16);
}

// --- per-wave 64x64x64 bf16 matmul: Ds = Ls @ B, B given as Bs[n][k] -------
__device__ __forceinline__ void mm64(const bf16_t* Ls, const bf16_t* Bs,
                                     bf16_t* Ds, int lane) {
#pragma unroll
    for (int mi = 0; mi < 4; ++mi) {
#pragma unroll
        for (int ni = 0; ni < 4; ++ni) {
            v8f acc = {};
#pragma unroll
            for (int kc = 0; kc < 2; ++kc) {
                v16bf a = lds_load_afrag(Ls, 64, mi * 16, kc * 32, lane);
                v16bf b = lds_load_bfrag(Bs, 64, ni * 16, kc * 32, lane);
                acc = __builtin_amdgcn_wmma_f32_16x16x32_bf16(
                    false, a, false, b, (short)0, acc, false, false);
            }
            int g = (lane >> 4) & 1, n = lane & 15;
#pragma unroll
            for (int r = 0; r < 8; ++r)
                Ds[(size_t)(mi * 16 + r + g * 8) * 64 + ni * 16 + n] =
                    (bf16_t)acc[r];
        }
    }
}

// transpose 64x64 bf16: out[j][i] = in[i][j]  (global row-major -> LDS)
__device__ __forceinline__ void load_transposed(const bf16_t* g, bf16_t* o, int lane) {
    for (int i = lane; i < 64; i += 32) {
        const bf16_t* row = g + (size_t)i * 64;
#pragma unroll
        for (int t = 0; t < 64; t += 8) {
            union { uint4 v; bf16_t e[8]; } u;
            u.v = *(const uint4*)(row + t);
#pragma unroll
            for (int j = 0; j < 8; ++j)
                o[(size_t)(t + j) * 64 + i] = u.e[j];
        }
    }
}
__device__ __forceinline__ void copy4096_g2l(const bf16_t* g, bf16_t* l, int lane) {
    for (int i = lane; i < 512; i += 32)
        *(uint4*)(l + (size_t)i * 8) = *(const uint4*)(g + (size_t)i * 8);
}
__device__ __forceinline__ void copy4096_l2g(const bf16_t* l, bf16_t* g, int lane) {
    for (int i = lane; i < 512; i += 32)
        *(uint4*)(g + (size_t)i * 8) = *(const uint4*)(l + (size_t)i * 8);
}
// Async global->LDS copy of one 8KB tile (GLOBAL_LOAD_ASYNC_TO_LDS_B128,
// tracked with ASYNCcnt).  16B per lane per issue, 16 issues per wave.
__device__ __forceinline__ void copy4096_g2l_async(const bf16_t* g, bf16_t* l, int lane) {
#pragma unroll
    for (int i = 0; i < 16; ++i) {
        int idx = (i * 32 + lane) * 8;
        unsigned lds_addr = (unsigned)(size_t)(l + idx);
        const bf16_t* ga = g + idx;
        asm volatile("global_load_async_to_lds_b128 %0, %1, off"
                     :: "v"(lds_addr), "v"(ga) : "memory");
    }
    asm volatile("s_wait_asynccnt 0x0" ::: "memory");
}

// ---------------------------------------------------------------------------
// 128x128-block WMMA GEMM, BK=64:  C = [gelu]( A @ B^T + bias [+ C] )
// A: (M,K) bf16 row-major.  Bt: (N,K) bf16 row-major.
// PERMCOL: permute output columns p -> (p&63)*64 + (p>>6)  (writes each
// 64x64 state matrix transposed so the scan can ingest tiles linearly).
// ---------------------------------------------------------------------------
template <bool GELU, bool BETA1, bool OUTBF16, bool PERMCOL>
__global__ __launch_bounds__(256) void gemm_bf16(
    const bf16_t* __restrict__ A, const bf16_t* __restrict__ Bt,
    const float* __restrict__ bias, void* __restrict__ Cout,
    int M, int N, int K) {
    __shared__ __align__(16) bf16_t As[128][64];
    __shared__ __align__(16) bf16_t Bs[128][64];   // n-major: Bs[n][k]
    int tid = threadIdx.x, lane = tid & 31, wave = tid >> 5;
    int m0 = blockIdx.y * 128, n0 = blockIdx.x * 128;
    int wm = wave * 16;
    v8f acc[8];
    v8f zero = {};
#pragma unroll
    for (int i = 0; i < 8; ++i) acc[i] = zero;

    for (int k0 = 0; k0 < K; k0 += 64) {
        {   // stage 128x64 A and B tiles (16KB each) via uint4 copies
            int r = tid >> 1, c = (tid & 1) * 32;         // 32 elems = 4 uint4
            const uint4* sa = (const uint4*)(A  + (size_t)(m0 + r) * K + k0 + c);
            const uint4* sb = (const uint4*)(Bt + (size_t)(n0 + r) * K + k0 + c);
            uint4* da = (uint4*)&As[r][c];
            uint4* db = (uint4*)&Bs[r][c];
#pragma unroll
            for (int j = 0; j < 4; ++j) da[j] = sa[j];
#pragma unroll
            for (int j = 0; j < 4; ++j) db[j] = sb[j];
            if (k0 + 64 < K) {
                __builtin_prefetch(sa + 8, 0, 0);          // global_prefetch
                __builtin_prefetch(sb + 8, 0, 0);
            }
        }
        __syncthreads();
#pragma unroll
        for (int kc = 0; kc < 2; ++kc) {
            v16bf a = lds_load_afrag(&As[0][0], 64, wm, kc * 32, lane);
#pragma unroll
            for (int ni = 0; ni < 8; ++ni) {
                v16bf b = lds_load_bfrag(&Bs[0][0], 64, ni * 16, kc * 32, lane);
                acc[ni] = __builtin_amdgcn_wmma_f32_16x16x32_bf16(
                    false, a, false, b, (short)0, acc[ni], false, false);
            }
        }
        __syncthreads();
    }
    int g = (lane >> 4) & 1, n = lane & 15;
#pragma unroll
    for (int ni = 0; ni < 8; ++ni) {
#pragma unroll
        for (int r = 0; r < 8; ++r) {
            int m = m0 + wm + r + g * 8;
            int p = n0 + ni * 16 + n;
            float v = acc[ni][r] + (bias ? bias[p] : 0.0f);
            size_t col = PERMCOL ? (size_t)((p & 63) * 64 + (p >> 6)) : (size_t)p;
            size_t idx = (size_t)m * N + col;
            if (BETA1) v += ((const float*)Cout)[idx];
            if (GELU)  v = gelu_f(v);
            if (OUTBF16) ((bf16_t*)Cout)[idx] = (bf16_t)v;
            else         ((float*)Cout)[idx]  = v;
        }
    }
}

// ---------------------------------------------------------------------------
// Chunked cumulative matrix-product scan (one wave per (batch, chunk)).
// Axb tiles are stored TRANSPOSED ([t][s]) by the Ax GEMM, so B-operand
// staging is a flat async copy; the running state stays row-major.
// ---------------------------------------------------------------------------
__global__ __launch_bounds__(32) void scan_phase1(const bf16_t* __restrict__ Ax,
                                                  bf16_t* __restrict__ G) {
    __shared__ __align__(16) bf16_t Bs[64 * 64], L0[64 * 64], L1[64 * 64];
    int lane = threadIdx.x & 31;
    int b = blockIdx.x >> 4, c = blockIdx.x & 15;
    const bf16_t* base = Ax + ((size_t)(b * CL + c * TCHUNK)) * 4096;
    load_transposed(base, L0, lane);      // stored [t][s] -> state [s][t]
    bf16_t* cur = L0; bf16_t* nxt = L1;
    for (int t = 1; t < TCHUNK; ++t) {
        __syncthreads();
        copy4096_g2l_async(base + (size_t)t * 4096, Bs, lane);
        __syncthreads();
        mm64(cur, Bs, nxt, lane);
        bf16_t* tmp = cur; cur = nxt; nxt = tmp;
    }
    __syncthreads();
    copy4096_l2g(cur, G + (size_t)(b * NCHUNK + c) * 4096, lane);
}

__global__ __launch_bounds__(32) void scan_phase2(const bf16_t* __restrict__ G,
                                                  bf16_t* __restrict__ E) {
    __shared__ __align__(16) bf16_t Bs[64 * 64], L0[64 * 64], L1[64 * 64];
    int lane = threadIdx.x & 31;
    int b = blockIdx.x;
    bf16_t* E0 = E + (size_t)b * NCHUNK * 4096;
    for (int i = lane; i < 512; i += 32) {      // E[0] = Identity
        union { uint4 v; bf16_t e[8]; } u;
#pragma unroll
        for (int j = 0; j < 8; ++j) {
            int idx = i * 8 + j;
            u.e[j] = ((idx >> 6) == (idx & 63)) ? (bf16_t)1.0f : (bf16_t)0.0f;
        }
        *(uint4*)(E0 + (size_t)i * 8) = u.v;
    }
    copy4096_g2l(G + (size_t)b * NCHUNK * 4096, L0, lane);   // G is row-major
    bf16_t* cur = L0; bf16_t* nxt = L1;
    for (int c = 1; c < NCHUNK; ++c) {
        __syncthreads();
        copy4096_l2g(cur, E + (size_t)(b * NCHUNK + c) * 4096, lane);
        load_transposed(G + (size_t)(b * NCHUNK + c) * 4096, Bs, lane);
        __syncthreads();
        mm64(cur, Bs, nxt, lane);
        bf16_t* tmp = cur; cur = nxt; nxt = tmp;
    }
}

__global__ __launch_bounds__(32) void scan_phase3(const bf16_t* __restrict__ Ax,
                                                  const bf16_t* __restrict__ E,
                                                  const float* __restrict__ pW,
                                                  const float* __restrict__ pb,
                                                  bf16_t* __restrict__ cpb) {
    __shared__ __align__(16) bf16_t Bs[64 * 64], L0[64 * 64], L1[64 * 64], Es[64 * 64];
    __shared__ float wv[64], vv[64];
    int lane = threadIdx.x & 31;
    int b = blockIdx.x >> 4, c = blockIdx.x & 15;
    copy4096_g2l(E + (size_t)(b * NCHUNK + c) * 4096, Es, lane);
    wv[lane] = pW[lane]; wv[lane + 32] = pW[lane + 32];
    float pbias = pb[0];
    const bf16_t* base = Ax + ((size_t)(b * CL + c * TCHUNK)) * 4096;
    load_transposed(base, L0, lane);
    bf16_t* cur = L0; bf16_t* nxt = L1;
    for (int t = 0; t < TCHUNK; ++t) {
        if (t > 0) {
            __syncthreads();
            copy4096_g2l_async(base + (size_t)t * 4096, Bs, lane);
            __syncthreads();
            mm64(cur, Bs, nxt, lane);
            bf16_t* tmp = cur; cur = nxt; nxt = tmp;
        }
        __syncthreads();
        int m0 = lane * 2;
        float s0 = 0.0f, s1 = 0.0f;            // v = L(t) @ w
#pragma unroll 8
        for (int k = 0; k < 64; ++k) {
            float wk = wv[k];
            s0 += (float)cur[(size_t)m0 * 64 + k] * wk;
            s1 += (float)cur[(size_t)(m0 + 1) * 64 + k] * wk;
        }
        vv[m0] = s0; vv[m0 + 1] = s1;
        __syncthreads();
        float o0 = pbias, o1 = pbias;          // cp = E @ v + pb
#pragma unroll 8
        for (int k = 0; k < 64; ++k) {
            float vk = vv[k];
            o0 += (float)Es[(size_t)m0 * 64 + k] * vk;
            o1 += (float)Es[(size_t)(m0 + 1) * 64 + k] * vk;
        }
        bf16x2 o; o.x = (bf16_t)o0; o.y = (bf16_t)o1;
        *(bf16x2*)(cpb + (size_t)(b * CL + c * TCHUNK + t) * 64 + m0) = o;
    }
}

// ---------------------------------------------------------------------------
__global__ __launch_bounds__(256) void embed_kernel(const int* __restrict__ tok,
                                                    const float* __restrict__ emb,
                                                    float* __restrict__ x,
                                                    bf16_t* __restrict__ xb) {
    int row = blockIdx.x, tid = threadIdx.x;
    int t = tok[row];
    const float* src = emb + (size_t)t * CDM;
    size_t off = (size_t)row * CDM;
    float v0 = src[tid], v1 = src[tid + 256];
    x[off + tid] = v0;           x[off + tid + 256] = v1;
    xb[off + tid] = (bf16_t)v0;  xb[off + tid + 256] = (bf16_t)v1;
}

__global__ __launch_bounds__(256) void add_ln_kernel(float* __restrict__ x,
                                                     bf16_t* __restrict__ xb,
                                                     const float* __restrict__ z,
                                                     const float* __restrict__ gw,
                                                     const float* __restrict__ gb) {
    __shared__ float s1[256], s2[256];
    int row = blockIdx.x, tid = threadIdx.x;
    size_t off = (size_t)row * CDM;
    float v0 = x[off + tid]       + z[off + tid];
    float v1 = x[off + tid + 256] + z[off + tid + 256];
    s1[tid] = v0 + v1;
    s2[tid] = v0 * v0 + v1 * v1;
    __syncthreads();
    for (int s = 128; s > 0; s >>= 1) {
        if (tid < s) { s1[tid] += s1[tid + s]; s2[tid] += s2[tid + s]; }
        __syncthreads();
    }
    float mean = s1[0] * (1.0f / CDM);
    float var  = s2[0] * (1.0f / CDM) - mean * mean;
    float rstd = rsqrtf(var + LNEPS);
    float o0 = (v0 - mean) * rstd * gw[tid]       + gb[tid];
    float o1 = (v1 - mean) * rstd * gw[tid + 256] + gb[tid + 256];
    x[off + tid] = o0;           x[off + tid + 256] = o1;
    xb[off + tid] = (bf16_t)o0;  xb[off + tid + 256] = (bf16_t)o1;
}

// elementwise f32 -> bf16 (n % 1024 == 0), 4 elems/thread
__global__ __launch_bounds__(256) void cvt_bf16_kernel(const float* __restrict__ in,
                                                       bf16_t* __restrict__ out) {
    int base = (blockIdx.x * 256 + threadIdx.x) * 4;
    float4 f = *(const float4*)(in + base);
    union { uint2 v; bf16_t e[4]; } u;
    u.e[0] = (bf16_t)f.x; u.e[1] = (bf16_t)f.y;
    u.e[2] = (bf16_t)f.z; u.e[3] = (bf16_t)f.w;
    *(uint2*)(out + base) = u.v;
}
// Aw (512 x 4096 f32) -> (4096 x 512 bf16) transposed   grid=4096, block=512
__global__ __launch_bounds__(512) void cvtT_bf16_kernel(const float* __restrict__ in,
                                                        bf16_t* __restrict__ out) {
    int n = blockIdx.x, k = threadIdx.x;
    out[(size_t)n * 512 + k] = (bf16_t)in[(size_t)k * 4096 + n];
}

// ---------------------------------------------------------------------------
extern "C" void kernel_launch(void* const* d_in, const int* in_sizes, int n_in,
                              void* d_out, int out_size, void* d_ws, size_t ws_size,
                              hipStream_t stream) {
    (void)in_sizes; (void)n_in; (void)out_size; (void)ws_size;
    const int*   tokens = (const int*)  d_in[0];
    const float* emb    = (const float*)d_in[1];
    const float* Aw     = (const float*)d_in[2];
    const float* projW  = (const float*)d_in[3];
    const float* projb  = (const float*)d_in[4];
    const float* CW     = (const float*)d_in[5];
    const float* Cb     = (const float*)d_in[6];
    const float* DW     = (const float*)d_in[7];
    const float* Db     = (const float*)d_in[8];
    const float* lnw    = (const float*)d_in[9];
    const float* lnb    = (const float*)d_in[10];
    const float* headW  = (const float*)d_in[11];
    const float* headb  = (const float*)d_in[12];

    char* w = (char*)d_ws;                              // workspace map:
    float*  x     = (float*)(w);                        //  16 MB f32 activations
    float*  z     = (float*)(w + 16 * MB);              //  16 MB f32 block output
    bf16_t* xb    = (bf16_t*)(w + 32 * MB);             //   8 MB bf16 activations
    bf16_t* cpb   = (bf16_t*)(w + 40 * MB);             //   1 MB bf16 scan output
    bf16_t* Axb   = (bf16_t*)(w + 41 * MB);             //  64 MB bf16 gelu(x@A), [t][s] tiles
    bf16_t* G     = (bf16_t*)(w + 105 * MB);            //   2 MB chunk products
    bf16_t* E     = (bf16_t*)(w + 107 * MB);            //   2 MB chunk prefixes
    bf16_t* AxWt  = (bf16_t*)(w + 109 * MB);            //  12 MB A weights (4096,512) bf16
    bf16_t* DWb   = (bf16_t*)(w + 121 * MB);            // 1.5 MB DW bf16
    bf16_t* CWb   = (bf16_t*)(w + 123 * MB);            // 0.2 MB CW bf16
    bf16_t* headWb= (bf16_t*)(w + 124 * MB);            //   1 MB head bf16

    // one-time weight conversion (bf16, B^T layouts)
    for (int i = 0; i < CNL; ++i)
        cvtT_bf16_kernel<<<4096, 512, 0, stream>>>(Aw + (size_t)i * CDM * 4096,
                                                   AxWt + (size_t)i * 4096 * CDM);
    cvt_bf16_kernel<<<(CNL * CDM * CDM) / 1024, 256, 0, stream>>>(DW, DWb);
    cvt_bf16_kernel<<<(CNL * CDM * CDS) / 1024, 256, 0, stream>>>(CW, CWb);
    cvt_bf16_kernel<<<(CV * CDM) / 1024, 256, 0, stream>>>(headW, headWb);

    embed_kernel<<<ROWS, 256, 0, stream>>>(tokens, emb, x, xb);

    for (int i = 0; i < CNL; ++i) {
        // Ax = gelu(x @ A_i): (8192x512)@(512x4096), bf16 out, transposed tiles
        gemm_bf16<true, false, true, true><<<dim3(4096 / 128, ROWS / 128), 256, 0, stream>>>(
            xb, AxWt + (size_t)i * 4096 * CDM, nullptr, Axb, ROWS, 4096, CDM);
        // chunked left-product scan + projection -> cpb
        scan_phase1<<<CB * NCHUNK, 32, 0, stream>>>(Axb, G);
        scan_phase2<<<CB, 32, 0, stream>>>(G, E);
        scan_phase3<<<CB * NCHUNK, 32, 0, stream>>>(Axb, E, projW + i * CDS,
                                                    projb + i, cpb);
        // z = x @ DW^T + Db ; z += cp @ CW^T + Cb
        gemm_bf16<false, false, false, false><<<dim3(CDM / 128, ROWS / 128), 256, 0, stream>>>(
            xb, DWb + (size_t)i * CDM * CDM, Db + i * CDM, z, ROWS, CDM, CDM);
        gemm_bf16<false, true, false, false><<<dim3(CDM / 128, ROWS / 128), 256, 0, stream>>>(
            cpb, CWb + (size_t)i * CDM * CDS, Cb + i * CDM, z, ROWS, CDM, CDS);
        // x = LayerNorm(x + z)  (also refreshes bf16 copy)
        add_ln_kernel<<<ROWS, 256, 0, stream>>>(x, xb, z, lnw + i * CDM, lnb + i * CDM);
    }
    // logits = x @ headW^T + headb -> d_out (f32)
    gemm_bf16<false, false, false, false><<<dim3(CV / 128, ROWS / 128), 256, 0, stream>>>(
        xb, headWb, headb, (float*)d_out, ROWS, CV, CDM);
}
